// ConvertedBlockLSTM_19490561589817
// MI455X (gfx1250) — matrix operand order repaired and checked by reading, not compiled, not run on hardware
//
#include <hip/hip_runtime.h>
#include <hip/hip_bf16.h>
#include <stdint.h>

typedef __attribute__((ext_vector_type(16))) _Float16 v16h;
typedef __attribute__((ext_vector_type(8)))  _Float16 v8h;
typedef __attribute__((ext_vector_type(8)))  float    v8f;

#define TSEQ     2048
#define NCELL    16     // blocks handled per workgroup (one batch element)
#define KDIM     64     // input size == hidden size per block
#define GDIM     256    // 4*BH gate outputs per cell
#define NW       4      // waves per workgroup
#define NTHREADS (NW * 32)
#define NT4      4      // N-tiles per wave (one per gate region)
#define CT       4      // timesteps per x-projection chunk
#define NCHUNK   (TSEQ / CT)

#define LOG2E 1.44269504088896f

// ---- branch-free activations on v_exp_f32 / v_rcp_f32 / v_tanh_f32 ----
#if __has_builtin(__builtin_amdgcn_exp2f)
#define EXP2F(x) __builtin_amdgcn_exp2f(x)
#else
#define EXP2F(x) exp2f(x)
#endif
#if __has_builtin(__builtin_amdgcn_rcpf)
#define RCPF(x) __builtin_amdgcn_rcpf(x)
#else
#define RCPF(x) (1.0f / (x))
#endif

__device__ __forceinline__ float fsigm(float x) {
  return RCPF(1.0f + EXP2F(-LOG2E * x));          // inf-safe: rcp(inf) = 0
}
#if __has_builtin(__builtin_amdgcn_tanhf)
__device__ __forceinline__ float ftanh(float x) { return __builtin_amdgcn_tanhf(x); }
#elif __has_builtin(__builtin_amdgcn_tanh_f32)
__device__ __forceinline__ float ftanh(float x) { return __builtin_amdgcn_tanh_f32(x); }
#else
__device__ __forceinline__ float ftanh(float x) {
  float e2x = EXP2F(2.0f * LOG2E * x);            // e^{2x}
  return 1.0f - 2.0f * RCPF(e2x + 1.0f);          // saturates to +-1 via inf/0
}
#endif

// ---- gfx1250 async global->LDS copies (ASYNCcnt) ----
__device__ __forceinline__ void async_b128(uint32_t lds, const void* g) {
  asm volatile("global_load_async_to_lds_b128 %0, %1, off" :: "v"(lds), "v"(g) : "memory");
}
__device__ __forceinline__ void wait_async0() {
  asm volatile("s_wait_asynccnt 0x0" ::: "memory");
}

// ---- A-matrix fragment builders (16-bit A 16x32 layout):
// lane<16 : M=lane,    halves 0..7 -> K=k0..k0+7,    halves 8..15 -> K=k0+16..k0+23
// lane>=16: M=lane-16, halves 0..7 -> K=k0+8..k0+15, halves 8..15 -> K=k0+24..k0+31
__device__ __forceinline__ v16h loadA_f16(const _Float16* src, int lane, int ks) {
  const int m  = lane & 15;
  const int kh = (lane >> 4) * 8;
  const _Float16* r = src + m * KDIM + ks * 32 + kh;
  v8h c0 = *(const v8h*)(r);
  v8h c1 = *(const v8h*)(r + 16);
  v16h a;
#pragma unroll
  for (int i = 0; i < 8; ++i) { a[i] = c0[i]; a[i + 8] = c1[i]; }
  return a;
}
__device__ __forceinline__ v16h loadA_f32(const float* src, int lane, int ks) {
  const int m  = lane & 15;
  const int kh = (lane >> 4) * 8;
  const float* r = src + m * KDIM + ks * 32 + kh;
  v16h a;
#pragma unroll
  for (int i = 0; i < 8; ++i) { a[i] = (_Float16)r[i]; a[i + 8] = (_Float16)r[16 + i]; }
  return a;
}

template<bool IN_F16, bool OUT_F16>
__global__ __launch_bounds__(NTHREADS)
void block_lstm_layer(const void* __restrict__ xin,
                      const float* __restrict__ w_ih,
                      const float* __restrict__ w_hh,
                      const float* __restrict__ b_ih,
                      const float* __restrict__ b_hh,
                      void* __restrict__ yout)
{
  // chunk staging for x (fp32 layer0 / f16 layer1), double buffered: 32KB
  __shared__ __align__(16) char     xstage[2][CT * NCELL * KDIM * 4];
  // x-projection (+bias) stored in raw WMMA D-layout, fp32: reloaded as the C
  // operand of the recurrent WMMA. [tl][wave][tile][lane*8+r] : 64KB
  __shared__ __align__(32) float    uxw[CT][NW][NT4][32 * 8];
  // double-buffered h (f16): removes the read/write hazard -> 1 barrier/step
  __shared__ __align__(16) _Float16 hbuf[2][NCELL * KDIM];

  const int tid  = threadIdx.x;
  const int lane = tid & 31;
  const int wv   = tid >> 5;          // wave w owns cols g = 64*gt + 16*w + col
  const int b    = blockIdx.x;
  const int col  = lane & 15;
  const int mh   = 8 * (lane >> 4);   // D-layout row base for this lane

#pragma unroll
  for (int i = 0; i < (NCELL * KDIM) / NTHREADS; ++i)
    hbuf[0][tid + NTHREADS * i] = (_Float16)0.0f;

  // per-lane gate bias, one per owned N-tile (folded into ux at store time)
  float bias[NT4];
#pragma unroll
  for (int gt = 0; gt < NT4; ++gt) {
    const int g = gt * 64 + 16 * wv + col;
    bias[gt] = b_ih[g] + b_hh[g];
  }

  // ---- per-wave weight tiles, converted to f16 once, resident in VGPRs ----
  // B-matrix 32x16 layout: lane<16 -> N=lane, K=kb..kb+7,kb+16..kb+23 ; lane>=16 -> N=lane-16, K=kb+8..kb+15,kb+24..kb+31
  v16h Bx[NT4][2], Bh[NT4][2];
  {
    const int kb = 8 * (lane >> 4);
#pragma unroll
    for (int gt = 0; gt < NT4; ++gt) {
      const int g = gt * 64 + 16 * wv + col;
#pragma unroll
      for (int ks = 0; ks < 2; ++ks) {
        const float* pi = w_ih + (size_t)g * KDIM + 32 * ks + kb;
        const float* ph = w_hh + (size_t)g * KDIM + 32 * ks + kb;
        v16h bi2, bh2;
#pragma unroll
        for (int i = 0; i < 16; ++i) { const int kk = i < 8 ? i : i + 8; bi2[i] = (_Float16)pi[kk]; bh2[i] = (_Float16)ph[kk]; }
        Bx[gt][ks] = bi2; Bh[gt][ks] = bh2;
      }
    }
  }

  const int    ESZ    = IN_F16 ? 2 : 4;
  const int    CBYTES = CT * NCELL * KDIM * ESZ;   // bytes per chunk (16KB / 8KB)
  const char*  xsrc   = (const char*)xin + (size_t)b * TSEQ * (NCELL * KDIM) * ESZ;

  // ---- async-prefetch chunk 0 into buffer 0 ----
  {
    const uint32_t l0 = (uint32_t)(uintptr_t)&xstage[0][0];
#pragma unroll
    for (int o = tid * 16; o < CBYTES; o += NTHREADS * 16)
      async_b128(l0 + o, xsrc + o);
  }
  wait_async0();
  __syncthreads();

  float c[8];
#pragma unroll
  for (int r = 0; r < 8; ++r) c[r] = 0.0f;

  for (int tc = 0; tc < NCHUNK; ++tc) {
    const int cur = tc & 1, nxt = cur ^ 1;

    // async-prefetch next chunk while we work on this one
    if (tc + 1 < NCHUNK) {
      const uint32_t ln = (uint32_t)(uintptr_t)&xstage[nxt][0];
      const char*    gp = xsrc + (size_t)(tc + 1) * CBYTES;
#pragma unroll
      for (int o = tid * 16; o < CBYTES; o += NTHREADS * 16)
        async_b128(ln + o, gp + o);
    }

    // ======== phase 1: batched x-projection for CT steps (off the h-chain) ========
#pragma unroll
    for (int tl = 0; tl < CT; ++tl) {
      const char* xrow = xstage[cur] + (size_t)tl * NCELL * KDIM * ESZ;
      v8f a[NT4] = {};
#pragma unroll
      for (int ks = 0; ks < 2; ++ks) {
        v16h A;
        if constexpr (IN_F16) A = loadA_f16((const _Float16*)xrow, lane, ks);
        else                  A = loadA_f32((const float*)xrow,   lane, ks);
#pragma unroll
        for (int gt = 0; gt < NT4; ++gt)
          a[gt] = __builtin_amdgcn_wmma_f32_16x16x32_f16(false, A, false, Bx[gt][ks], (short)0, a[gt], false, false);
      }
#pragma unroll
      for (int gt = 0; gt < NT4; ++gt) {
        v8f o = a[gt];
#pragma unroll
        for (int r = 0; r < 8; ++r) o[r] += bias[gt];
        *(v8f*)&uxw[tl][wv][gt][lane * 8] = o;     // raw D-layout, reused as WMMA C
      }
    }
    __syncthreads();   // ux ready for the whole chunk

    // ======== phase 2: recurrent scan — 1 barrier/step, gates in registers ========
    for (int tl = 0; tl < CT; ++tl) {
      const int t  = tc * CT + tl;
      const int rb = t & 1, wb = rb ^ 1;

      // accumulators initialized from x-projection (C operand), then h-GEMM
      v8f a[NT4];
#pragma unroll
      for (int gt = 0; gt < NT4; ++gt)
        a[gt] = *(const v8f*)&uxw[tl][wv][gt][lane * 8];
#pragma unroll
      for (int ks = 0; ks < 2; ++ks) {
        v16h A = loadA_f16(hbuf[rb], lane, ks);
#pragma unroll
        for (int gt = 0; gt < NT4; ++gt)
          a[gt] = __builtin_amdgcn_wmma_f32_16x16x32_f16(false, A, false, Bh[gt][ks], (short)0, a[gt], false, false);
      }

      // gates straight out of the D registers: lane owns (cells mh..mh+7, j = 16*wv+col)
      const int j = 16 * wv + col;
#pragma unroll
      for (int r = 0; r < 8; ++r) {
        const float i_t = fsigm(a[0][r]);
        const float f_t = fsigm(a[1][r]);
        const float g_t = ftanh(a[2][r]);
        const float o_t = fsigm(a[3][r]);
        const float cc  = f_t * c[r] + i_t * g_t;
        c[r] = cc;
        const float h = o_t * ftanh(cc);
        const int   e = (r + mh) * KDIM + j;       // cell*64 + j
        hbuf[wb][e] = (_Float16)h;
        const size_t oidx = ((size_t)b * TSEQ + t) * (NCELL * KDIM) + e;
        if constexpr (OUT_F16) ((_Float16*)yout)[oidx] = (_Float16)h;
        else                   ((float*)yout)[oidx]    = h;
      }
      __syncthreads();   // h_t visible to all waves for step t+1
    }

    wait_async0();       // next chunk landed in LDS (precedes barrier for visibility)
    __syncthreads();     // safe to switch xstage buffers / overwrite uxw
  }
}

extern "C" void kernel_launch(void* const* d_in, const int* in_sizes, int n_in,
                              void* d_out, int out_size, void* d_ws, size_t ws_size,
                              hipStream_t stream) {
  const float* x      = (const float*)d_in[0];
  const float* w_ih_0 = (const float*)d_in[1];
  const float* w_hh_0 = (const float*)d_in[2];
  const float* b_ih_0 = (const float*)d_in[3];
  const float* b_hh_0 = (const float*)d_in[4];
  const float* w_ih_1 = (const float*)d_in[5];
  const float* w_hh_1 = (const float*)d_in[6];
  const float* b_ih_1 = (const float*)d_in[7];
  const float* b_hh_1 = (const float*)d_in[8];

  _Float16* inter = (_Float16*)d_ws;   // layer-0 output, f16, B*T*1024 elements (67 MB)

  // Layer 0: fp32 x in, f16 intermediate out
  block_lstm_layer<false, true><<<16, NTHREADS, 0, stream>>>(
      x, w_ih_0, w_hh_0, b_ih_0, b_hh_0, inter);
  // Layer 1: f16 intermediate in, fp32 final out
  block_lstm_layer<true, false><<<16, NTHREADS, 0, stream>>>(
      inter, w_ih_1, w_hh_1, b_ih_1, b_hh_1, (float*)d_out);
}
